// GNNRegressionWithSkipConnections_84301618086373
// MI455X (gfx1250) — matrix operand order repaired and checked by reading, not compiled, not run on hardware
//
#include <hip/hip_runtime.h>
#include <hip/hip_bf16.h>

// Problem constants (fixed by the reference)
#define Nn 50000
#define Ee 1600000
#define Hh 128
#define MROWS 5   // m-tiles per wave in the GEMM (3125 = 625 * 5, exact)

typedef __attribute__((ext_vector_type(16))) __bf16 v16bf;
typedef __attribute__((ext_vector_type(8)))  __bf16 v8bf;
typedef __attribute__((ext_vector_type(8)))  float  v8f;

__device__ __forceinline__ float mishf(float x) {
  float sp = (x > 20.0f) ? x : log1pf(__expf(x));
  return x * tanhf(sp);
}

// ---------------- degree / normalization ----------------
__global__ void k_init_deg(float* __restrict__ deg) {
  int i = blockIdx.x * blockDim.x + threadIdx.x;
  if (i < Nn) deg[i] = 1.0f;  // self-loop contributes 1
}

__global__ void k_deg_accum(const long long* __restrict__ ei, float* __restrict__ deg) {
  int e = blockIdx.x * blockDim.x + threadIdx.x;
  if (e < Ee) atomicAdd(&deg[(int)ei[(size_t)Ee + e]], 1.0f);
}

__global__ void k_rsqrt_inplace(float* __restrict__ deg) {
  int i = blockIdx.x * blockDim.x + threadIdx.x;
  if (i < Nn) deg[i] = rsqrtf(deg[i]);  // deg >= 1 always
}

// ---------------- one-time weight prep: split + transpose to [n][k] bf16 -------
__global__ void k_wsplit_t(const float* __restrict__ W,
                           __bf16* __restrict__ hi, __bf16* __restrict__ lo) {
  int idx = blockIdx.x * blockDim.x + threadIdx.x;  // 128*128
  if (idx >= Hh * Hh) return;
  int k = idx >> 7, n = idx & 127;
  float  f  = W[idx];                 // W[k][n], row-major
  __bf16 hb = (__bf16)f;
  hi[n * Hh + k] = hb;                // transposed: [n][k]
  lo[n * Hh + k] = (__bf16)(f - (float)hb);
}

// ---------------- activation split: f32 -> bf16 hi/lo --------------------------
__global__ void k_split(const float* __restrict__ in,
                        __bf16* __restrict__ hi, __bf16* __restrict__ lo, int n) {
  int i = blockIdx.x * blockDim.x + threadIdx.x;
  if (i >= n) return;
  float  f  = in[i];
  __bf16 hb = (__bf16)f;
  hi[i] = hb;
  lo[i] = (__bf16)(f - (float)hb);
}

// ---------------- WMMA GEMM: C[N,128] = A[N,128] @ W[128,128] ------------------
// A pre-split to bf16 hi/lo (row-major); W pre-split + transposed to [n][k].
// 3-product bf16 split: C += Ah*Bh + Ah*Bl + Al*Bh  (near-fp32 accuracy).
// One wave owns a 16-column tile; B fragments live in registers across MROWS
// m-tiles. Inner loop is pure b128 loads + v_wmma.
// mode 0: Cf[m][n] = acc + bias1 + bias2 (f32)
// mode 1: mish(acc + bias) split-stored to Ohi/Olo (bf16 pair)
__global__ void k_gemm_wmma(const __bf16* __restrict__ Ahi,
                            const __bf16* __restrict__ Alo,
                            const __bf16* __restrict__ Bhi,
                            const __bf16* __restrict__ Blo,
                            const float* __restrict__ bias1,   // nullable
                            const float* __restrict__ bias2,   // nullable
                            float* __restrict__ Cf,
                            __bf16* __restrict__ Ohi,
                            __bf16* __restrict__ Olo,
                            int mode) {
  const int wave  = (blockIdx.x * blockDim.x + threadIdx.x) >> 5;
  const int lane  = threadIdx.x & 31;
  const int half  = lane >> 4;         // 0: lanes 0-15, 1: lanes 16-31
  const int lr    = lane & 15;
  const int ntile = wave & 7;
  const int mgrp  = wave >> 3;
  const int ncol  = ntile * 16 + lr;

  // Load B fragments once (K in {kb..kb+7} U {kb+16..kb+23}, kb = kt*32 + half*8, [n][k] layout)
  v16bf bh[4], bl[4];
  const __bf16* brow_h = Bhi + (size_t)ncol * Hh;
  const __bf16* brow_l = Blo + (size_t)ncol * Hh;
#pragma unroll
  for (int kt = 0; kt < 4; ++kt) {
    const int kb = kt * 32 + half * 8;
    v8bf b0 = *(const v8bf*)(brow_h + kb);
    v8bf b1 = *(const v8bf*)(brow_h + kb + 16);
    bh[kt] = __builtin_shufflevector(b0, b1, 0,1,2,3,4,5,6,7,8,9,10,11,12,13,14,15);
    v8bf c0 = *(const v8bf*)(brow_l + kb);
    v8bf c1 = *(const v8bf*)(brow_l + kb + 16);
    bl[kt] = __builtin_shufflevector(c0, c1, 0,1,2,3,4,5,6,7,8,9,10,11,12,13,14,15);
  }

  float bb = 0.0f;
  if (bias1) bb += bias1[ncol];
  if (bias2) bb += bias2[ncol];

#pragma unroll
  for (int mt = 0; mt < MROWS; ++mt) {
    const int m0 = (mgrp * MROWS + mt) << 4;
    const __bf16* arow_h = Ahi + (size_t)(m0 + lr) * Hh;
    const __bf16* arow_l = Alo + (size_t)(m0 + lr) * Hh;
    if (mt + 1 < MROWS) {
      __builtin_prefetch(arow_h + 16 * Hh, 0, 0);
      __builtin_prefetch(arow_l + 16 * Hh, 0, 0);
    }
    v8f acc = {};
#pragma unroll
    for (int kt = 0; kt < 4; ++kt) {
      const int k0 = kt * 32 + half * 8;   // A: K in {k0..k0+7} U {k0+16..k0+23}
      v8bf a0 = *(const v8bf*)(arow_h + k0);
      v8bf a1 = *(const v8bf*)(arow_h + k0 + 16);
      v16bf ah = __builtin_shufflevector(a0, a1, 0,1,2,3,4,5,6,7,8,9,10,11,12,13,14,15);
      v8bf a2 = *(const v8bf*)(arow_l + k0);
      v8bf a3 = *(const v8bf*)(arow_l + k0 + 16);
      v16bf al = __builtin_shufflevector(a2, a3, 0,1,2,3,4,5,6,7,8,9,10,11,12,13,14,15);
      acc = __builtin_amdgcn_wmma_f32_16x16x32_bf16(false, ah, false, bh[kt], (short)0, acc, false, false);
      acc = __builtin_amdgcn_wmma_f32_16x16x32_bf16(false, ah, false, bl[kt], (short)0, acc, false, false);
      acc = __builtin_amdgcn_wmma_f32_16x16x32_bf16(false, al, false, bh[kt], (short)0, acc, false, false);
    }

    // C/D layout: lane -> col ncol, rows m0 + 8*half + i (i=0..7)
    const size_t base = (size_t)(m0 + half * 8) * Hh + ncol;
    if (mode == 0) {
#pragma unroll
      for (int i = 0; i < 8; ++i) Cf[base + (size_t)i * Hh] = acc[i] + bb;
    } else {
#pragma unroll
      for (int i = 0; i < 8; ++i) {
        float  v  = mishf(acc[i] + bb);
        __bf16 hb = (__bf16)v;
        Ohi[base + (size_t)i * Hh] = hb;
        Olo[base + (size_t)i * Hh] = (__bf16)(v - (float)hb);
      }
    }
  }
}

// ---------------- edge aggregation: agg[dst] += t[src] * dinv[src]*dinv[dst] ----
__global__ void k_edge_agg(const long long* __restrict__ ei,
                           const float* __restrict__ dinv,
                           const float* __restrict__ t,
                           float* __restrict__ agg) {
  long long tid = (long long)blockIdx.x * blockDim.x + threadIdx.x;
  int e = (int)(tid >> 5);
  if (e >= Ee) return;
  int lane = (int)(tid & 31);
  int s = (int)ei[e];
  int d = (int)ei[(size_t)Ee + e];
  float nrm = dinv[s] * dinv[d];
  const float4 v = *(const float4*)(t + (size_t)s * Hh + lane * 4);
  float* o = agg + (size_t)d * Hh + lane * 4;
  atomicAdd(o + 0, v.x * nrm);
  atomicAdd(o + 1, v.y * nrm);
  atomicAdd(o + 2, v.z * nrm);
  atomicAdd(o + 3, v.w * nrm);
}

// ---------------- h = mish(agg + dinv^2 * t), split-stored as bf16 hi/lo -------
__global__ void k_mish_combine(const float* __restrict__ agg,
                               const float* __restrict__ t,
                               const float* __restrict__ dinv,
                               __bf16* __restrict__ hhi, __bf16* __restrict__ hlo) {
  int i = blockIdx.x * blockDim.x + threadIdx.x;
  if (i >= Nn * Hh) return;
  int n = i >> 7;
  float di = dinv[n];
  float  v  = mishf(agg[i] + di * di * t[i]);
  __bf16 hb = (__bf16)v;
  hhi[i] = hb;
  hlo[i] = (__bf16)(v - (float)hb);
}

// ---------------- out[i] = dot(h[i,:], W_post) + b_post  (wave per node) -------
__global__ void k_post(const __bf16* __restrict__ hhi, const __bf16* __restrict__ hlo,
                       const float* __restrict__ Wp, const float* __restrict__ bp,
                       float* __restrict__ out) {
  int node = blockIdx.x * 8 + (threadIdx.x >> 5);
  int lane = threadIdx.x & 31;
  if (node >= Nn) return;
  const size_t base = (size_t)node * Hh + lane * 4;
  const float4 wv = *(const float4*)(Wp + lane * 4);
  float h0 = (float)hhi[base + 0] + (float)hlo[base + 0];
  float h1 = (float)hhi[base + 1] + (float)hlo[base + 1];
  float h2 = (float)hhi[base + 2] + (float)hlo[base + 2];
  float h3 = (float)hhi[base + 3] + (float)hlo[base + 3];
  float s = h0 * wv.x + h1 * wv.y + h2 * wv.z + h3 * wv.w;
#pragma unroll
  for (int off = 16; off > 0; off >>= 1) s += __shfl_xor(s, off, 32);
  if (lane == 0) out[node] = s + bp[0];
}

extern "C" void kernel_launch(void* const* d_in, const int* in_sizes, int n_in,
                              void* d_out, int out_size, void* d_ws, size_t ws_size,
                              hipStream_t stream) {
  const float*     x      = (const float*)d_in[0];
  const long long* ei     = (const long long*)d_in[1];
  const float*     W_pre  = (const float*)d_in[2];
  const float*     b_pre  = (const float*)d_in[3];
  const float*     gcn_W  = (const float*)d_in[4];
  const float*     gcn_b  = (const float*)d_in[5];
  const float*     skip_W = (const float*)d_in[6];
  const float*     skip_b = (const float*)d_in[7];
  const float*     W_post = (const float*)d_in[8];
  const float*     b_post = (const float*)d_in[9];
  float* out = (float*)d_out;

  // Workspace layout (~77.5 MB, hot set L2-resident):
  //   dinv[N] f32 | t[N,H] f32 | agg[N,H] f32 | h_hi[N,H] bf16 | h_lo[N,H] bf16
  //   | w_hi[7,H,H] bf16 | w_lo[7,H,H] bf16
  char*   ws   = (char*)d_ws;
  size_t  off  = 0;
  float*  dinv = (float*)(ws + off);  off += (size_t)Nn * 4;            off = (off + 1023) & ~(size_t)1023;
  float*  t    = (float*)(ws + off);  off += (size_t)Nn * Hh * 4;       off = (off + 1023) & ~(size_t)1023;
  float*  agg  = (float*)(ws + off);  off += (size_t)Nn * Hh * 4;       off = (off + 1023) & ~(size_t)1023;
  __bf16* hhi  = (__bf16*)(ws + off); off += (size_t)Nn * Hh * 2;       off = (off + 1023) & ~(size_t)1023;
  __bf16* hlo  = (__bf16*)(ws + off); off += (size_t)Nn * Hh * 2;       off = (off + 1023) & ~(size_t)1023;
  __bf16* whi  = (__bf16*)(ws + off); off += (size_t)7 * Hh * Hh * 2;   off = (off + 1023) & ~(size_t)1023;
  __bf16* wlo  = (__bf16*)(ws + off);

  // x's bf16 split transiently borrows the t buffer (t is dead until layer 0)
  __bf16* xhi = (__bf16*)t;
  __bf16* xlo = xhi + (size_t)Nn * Hh;

  // degrees -> dinv
  k_init_deg<<<(Nn + 255) / 256, 256, 0, stream>>>(dinv);
  k_deg_accum<<<(Ee + 255) / 256, 256, 0, stream>>>(ei, dinv);
  k_rsqrt_inplace<<<(Nn + 255) / 256, 256, 0, stream>>>(dinv);

  // one-time weight prep: slot 0 = W_pre, 1..3 = gcn_W, 4..6 = skip_W
  const int wgrid = (Hh * Hh + 255) / 256;
  k_wsplit_t<<<wgrid, 256, 0, stream>>>(W_pre, whi, wlo);
  for (int l = 0; l < 3; ++l) {
    k_wsplit_t<<<wgrid, 256, 0, stream>>>(gcn_W + (size_t)l * Hh * Hh,
                                          whi + (size_t)(1 + l) * Hh * Hh,
                                          wlo + (size_t)(1 + l) * Hh * Hh);
    k_wsplit_t<<<wgrid, 256, 0, stream>>>(skip_W + (size_t)l * Hh * Hh,
                                          whi + (size_t)(4 + l) * Hh * Hh,
                                          wlo + (size_t)(4 + l) * Hh * Hh);
  }

  const int gemm_blocks = 8 * (Nn / 16 / MROWS) / 4;  // 5000 waves / 4 per block = 1250

  // preprocess: h = mish(x @ W_pre + b_pre)  (split-stored)
  k_split<<<(Nn * Hh + 255) / 256, 256, 0, stream>>>(x, xhi, xlo, Nn * Hh);
  k_gemm_wmma<<<gemm_blocks, 128, 0, stream>>>(xhi, xlo, whi, wlo,
                                               b_pre, nullptr, nullptr, hhi, hlo, 1);

  for (int l = 0; l < 3; ++l) {
    // t = h @ gcn_W[l]
    k_gemm_wmma<<<gemm_blocks, 128, 0, stream>>>(
        hhi, hlo, whi + (size_t)(1 + l) * Hh * Hh, wlo + (size_t)(1 + l) * Hh * Hh,
        nullptr, nullptr, t, nullptr, nullptr, 0);
    // agg := h @ skip_W[l] + skip_b[l] + gcn_b[l]
    k_gemm_wmma<<<gemm_blocks, 128, 0, stream>>>(
        hhi, hlo, whi + (size_t)(4 + l) * Hh * Hh, wlo + (size_t)(4 + l) * Hh * Hh,
        skip_b + (size_t)l * Hh, gcn_b + (size_t)l * Hh, agg, nullptr, nullptr, 0);
    // scatter-add messages over edges (atomics resolve in L2)
    long long ethreads = (long long)Ee * 32;
    k_edge_agg<<<(int)((ethreads + 255) / 256), 256, 0, stream>>>(ei, dinv, t, agg);
    // h = mish(agg + dinv^2 * t), split-stored for next layer's GEMMs
    k_mish_combine<<<(Nn * Hh + 255) / 256, 256, 0, stream>>>(agg, t, dinv, hhi, hlo);
  }

  // postprocess: out = h @ W_post + b_post
  k_post<<<(Nn + 7) / 8, 256, 0, stream>>>(hhi, hlo, W_post, b_post, out);
}
